// Decoder_4380866642486
// MI455X (gfx1250) — matrix operand order repaired and checked by reading, not compiled, not run on hardware
//
#include <hip/hip_runtime.h>

// ---------------------------------------------------------------------------
// MI455X (gfx1250) Tacotron2-style decoder.
// Strategy (compile-only, reasoned from HW specs):
//  * All GEMMs via v_wmma_f32_16x16x32_f16 (wave32, 16x16 tiles, fp32 acc).
//  * Weights packed ONCE to f16 in WMMA B-fragment order (~29MB -> resident
//    in 192MB L2 across all 500 recurrent steps; HBM traffic negligible).
//  * Activations f16, recurrence state fp32 (VALU pointwise sig/tanh/zoneout).
//  * 500 sequential steps, each a small DAG of kernels on `stream`
//    (graph-capture safe: no malloc/sync, deterministic).
// ---------------------------------------------------------------------------

typedef __attribute__((ext_vector_type(16))) _Float16 v16h;
typedef __attribute__((ext_vector_type(8)))  _Float16 v8h;
typedef __attribute__((ext_vector_type(8)))  float    v8f;

#define ZONEOUT 0.1f
#define B_   64
#define TENC 200
#define ENC_ 256
#define DEC_ 1024
#define HATT 128
#define CONVC 32
#define KCONV 31
#define PRE_ 256
#define CCH  80
#define SSTEPS 500

__device__ __forceinline__ float sigf(float x) { return 1.0f / (1.0f + expf(-x)); }

// ---------------------------------------------------------------------------
// Generic WMMA GEMM: C[M,N] = A[M,K] * Bpacked[K,N], A f16 row-major (lda),
// B pre-packed in fragment order, C fp32 (ldc). M%16==0, N%64==0, K%32==0.
// One wave -> 16x64 output strip (4 WMMA tiles reusing the A fragment).
// ---------------------------------------------------------------------------
__global__ void gemm_wmma(const _Float16* __restrict__ A, int lda,
                          const _Float16* __restrict__ Bp,
                          float* __restrict__ C, int ldc,
                          int M, int N, int K) {
  const int lane = threadIdx.x & 31;
  const int wave = (blockIdx.x * blockDim.x + threadIdx.x) >> 5;
  const int mtiles = M >> 4;
  const int mt = wave % mtiles;
  const int ng = wave / mtiles;
  const int n0 = ng << 6;              // 4 n-tiles of 16
  if (n0 >= N) return;
  const int ktiles = K >> 5;
  const int hi = (lane & 16) ? 8 : 0;  // lanes 16..31 hold the +16 K-chunk
  const _Float16* arow = A + (size_t)(mt * 16 + (lane & 15)) * lda;
  const int nt0 = n0 >> 4;

  v8f acc[4] = {v8f{}, v8f{}, v8f{}, v8f{}};
  for (int kt = 0; kt < ktiles; ++kt) {
    // A fragment, contiguous K order matching packed B:
    //   halves 0..15 = K[kt*32 + 2*hi .. +15]
    const _Float16* ap = arow + kt * 32;
    v8h a0 = *(const v8h*)(ap + 2 * hi);
    v8h a1 = *(const v8h*)(ap + 2 * hi + 8);
    v16h afrag = __builtin_shufflevector(a0, a1, 0,1,2,3,4,5,6,7,8,9,10,11,12,13,14,15);
#pragma unroll
    for (int j = 0; j < 4; ++j) {
      const _Float16* bp = Bp + (((size_t)(nt0 + j) * ktiles + kt) * 32 + lane) * 16;
      __builtin_prefetch(bp + 512);          // next K tile -> global_prefetch_b8
      v16h bfrag = *(const v16h*)bp;         // 16 contiguous halves per lane
      acc[j] = __builtin_amdgcn_wmma_f32_16x16x32_f16(
          false, afrag, false, bfrag, (short)0, acc[j], false, false);
    }
  }
  // C/D layout: VGPR r -> row mt*16 + r + (lane<16?0:8), col = n-tile + lane%16
  const int rbase = mt * 16 + ((lane & 16) ? 8 : 0);
  const int cl = lane & 15;
#pragma unroll
  for (int j = 0; j < 4; ++j) {
    float* cp = C + (size_t)rbase * ldc + (n0 + j * 16 + cl);
#pragma unroll
    for (int r = 0; r < 8; ++r) cp[(size_t)r * ldc] = acc[j][r];
  }
}

// ---------------------------------------------------------------------------
// Weight packing into the B-fragment order assumed above:
//   dst[((nt*ktiles+kt)*32 + lane)*16 + j], n = nt*16 + lane%16,
//   k = kt*32 + (lane&16 ? 16:0) + j.  B[k][n] = W[n][k] (W is (N,K) row-major).
// ---------------------------------------------------------------------------
__global__ void k_pack_catK(const float* __restrict__ W1, const float* __restrict__ W2,
                            _Float16* __restrict__ dst, int N, int K1, int K) {
  size_t i = (size_t)blockIdx.x * blockDim.x + threadIdx.x;
  size_t total = (size_t)N * K;
  if (i >= total) return;
  int j = i & 15; int lane = (i >> 4) & 31; size_t tile = i >> 9;
  int ktiles = K >> 5;
  int kt = (int)(tile % ktiles); int nt = (int)(tile / ktiles);
  int n = nt * 16 + (lane & 15);
  int k = kt * 32 + ((lane & 16) ? 16 : 0) + j;
  float v = (k < K1) ? W1[(size_t)n * K1 + k]
                     : W2[(size_t)n * (K - K1) + (k - K1)];
  dst[i] = (_Float16)v;
}

__global__ void k_pack_catN(const float* __restrict__ W1, const float* __restrict__ W2,
                            _Float16* __restrict__ dst, int N1, int N2, int Npad, int K) {
  size_t i = (size_t)blockIdx.x * blockDim.x + threadIdx.x;
  size_t total = (size_t)Npad * K;
  if (i >= total) return;
  int j = i & 15; int lane = (i >> 4) & 31; size_t tile = i >> 9;
  int ktiles = K >> 5;
  int kt = (int)(tile % ktiles); int nt = (int)(tile / ktiles);
  int n = nt * 16 + (lane & 15);
  int k = kt * 32 + ((lane & 16) ? 16 : 0) + j;
  float v = 0.0f;
  if (n < N1) v = W1[(size_t)n * K + k];
  else if (n < N1 + N2) v = W2[(size_t)(n - N1) * K + k];
  dst[i] = (_Float16)v;
}

__global__ void k_f32_to_f16(const float* __restrict__ src, _Float16* __restrict__ dst, size_t n) {
  size_t i = (size_t)blockIdx.x * blockDim.x + threadIdx.x;
  if (i < n) dst[i] = (_Float16)src[i];
}
__global__ void k_zero_f32(float* p, size_t n) {
  size_t i = (size_t)blockIdx.x * blockDim.x + threadIdx.x;
  if (i < n) p[i] = 0.0f;
}
__global__ void k_zero_f16(_Float16* p, size_t n) {
  size_t i = (size_t)blockIdx.x * blockDim.x + threadIdx.x;
  if (i < n) p[i] = (_Float16)0.0f;
}

// ---------------------------------------------------------------------------
// Location-sensitive attention: one block per batch row.
// conv(att_prev) -> tanh(conv·Watt + pm + enc_b + h0·Wdec) · ww -> masked
// softmax -> context.  Writes aw to d_out att section, acc update, ac (f16)
// into LSTM0 input and hcs buffers.
// ---------------------------------------------------------------------------
__global__ void k_attention(const float* __restrict__ enc, const int* __restrict__ text_len,
                            const float* __restrict__ pm, const float* __restrict__ gdec,
                            const float* __restrict__ convw, const float* __restrict__ watt,
                            const float* __restrict__ ww, const float* __restrict__ wb,
                            const float* __restrict__ encb,
                            float* __restrict__ att_acc, float* __restrict__ att_out,
                            _Float16* __restrict__ A0, _Float16* __restrict__ Ahcs, int s) {
  __shared__ float s_ap[TENC + 2 * 15 + 2];
  __shared__ float s_watt[HATT * CONVC];
  __shared__ float s_cw[CONVC * KCONV];
  __shared__ float s_ww[HATT];
  __shared__ float s_g[HATT];
  __shared__ float s_e[TENC];
  __shared__ float s_red[256];
  const int b = blockIdx.x, tid = threadIdx.x;
  const int len = text_len[b];

  for (int i = tid; i < HATT * CONVC; i += 256) s_watt[i] = watt[i];
  for (int i = tid; i < CONVC * KCONV; i += 256) s_cw[i] = convw[i];
  for (int i = tid; i < HATT; i += 256) { s_ww[i] = ww[i]; s_g[i] = gdec[b * HATT + i] + encb[i]; }
  for (int i = tid; i < TENC + 32; i += 256) s_ap[i] = 0.0f;
  __syncthreads();
  float invlen = 1.0f / (float)len;
  for (int t = tid; t < TENC; t += 256)
    s_ap[15 + t] = (s == 0) ? ((t < len) ? invlen : 0.0f) : att_acc[b * TENC + t];
  __syncthreads();

  for (int t = tid; t < TENC; t += 256) {
    float conv[CONVC];
#pragma unroll
    for (int c = 0; c < CONVC; ++c) conv[c] = 0.0f;
    for (int j = 0; j < KCONV; ++j) {
      float av = s_ap[t + j];
#pragma unroll
      for (int c = 0; c < CONVC; ++c) conv[c] += s_cw[c * KCONV + j] * av;
    }
    float e = wb[0];
    const float* pmrow = pm + ((size_t)b * TENC + t) * HATT;
    for (int h = 0; h < HATT; ++h) {
      float x = s_g[h] + pmrow[h];
#pragma unroll
      for (int c = 0; c < CONVC; ++c) x += conv[c] * s_watt[h * CONVC + c];
      e += tanhf(x) * s_ww[h];
    }
    s_e[t] = (t < len) ? e : -__builtin_inff();
  }
  __syncthreads();
  // masked softmax over t
  float m = -3.4e38f;
  for (int t = tid; t < TENC; t += 256) m = fmaxf(m, s_e[t]);
  s_red[tid] = m; __syncthreads();
  for (int w = 128; w > 0; w >>= 1) { if (tid < w) s_red[tid] = fmaxf(s_red[tid], s_red[tid + w]); __syncthreads(); }
  float mx = s_red[0]; __syncthreads();
  float psum = 0.0f;
  for (int t = tid; t < TENC; t += 256) { float ex = expf(s_e[t] - mx); s_e[t] = ex; psum += ex; }
  s_red[tid] = psum; __syncthreads();
  for (int w = 128; w > 0; w >>= 1) { if (tid < w) s_red[tid] += s_red[tid + w]; __syncthreads(); }
  float inv = 1.0f / s_red[0]; __syncthreads();
  for (int t = tid; t < TENC; t += 256) {
    float wgt = s_e[t] * inv;
    s_e[t] = wgt;
    att_out[((size_t)b * SSTEPS + s) * TENC + t] = wgt;
    att_acc[b * TENC + t] = (s == 0) ? wgt : att_acc[b * TENC + t] + wgt;
  }
  __syncthreads();
  // context vector: one channel per thread (256 == ENC)
  const int kch = tid;
  float accv = 0.0f;
  const float* eb = enc + (size_t)b * TENC * ENC_ + kch;
  for (int t = 0; t < TENC; ++t) accv += s_e[t] * eb[(size_t)t * ENC_];
  A0[(size_t)b * 1536 + kch] = (_Float16)accv;
  Ahcs[(size_t)b * 1280 + 1024 + kch] = (_Float16)accv;
}

// prenet input for step s: zeros (go frame) at s==0, else teacher-forced frame.
__global__ void k_prenet_in(const float* __restrict__ ft, _Float16* __restrict__ pin, int s) {
  int i = blockIdx.x * blockDim.x + threadIdx.x;
  if (i >= B_ * 160) return;
  int b = i / 160, j = i % 160;
  float v = 0.0f;
  if (s > 0) v = ft[(size_t)b * CCH * 1000 + (size_t)(j % CCH) * 1000 + (s - 1) * 2 + j / CCH];
  pin[(size_t)b * 160 + j] = (_Float16)v;
}

__global__ void k_bias_relu_f16(const float* __restrict__ g, const float* __restrict__ bias,
                                _Float16* __restrict__ dst, int cols, int ldd) {
  int i = blockIdx.x * blockDim.x + threadIdx.x;
  if (i >= B_ * cols) return;
  int b = i / cols, c = i % cols;
  float v = fmaxf(g[(size_t)b * cols + c] + bias[c], 0.0f);
  dst[(size_t)b * ldd + c] = (_Float16)v;
}

__global__ void k_lstm_pw(const float* __restrict__ g, const float* __restrict__ bih,
                          const float* __restrict__ bhh, float* __restrict__ h,
                          float* __restrict__ c, _Float16* __restrict__ dsta, int ldda,
                          _Float16* __restrict__ dstb, int lddb) {
  int idx = blockIdx.x * blockDim.x + threadIdx.x;   // B_*DEC_
  if (idx >= B_ * DEC_) return;
  int b = idx >> 10, u = idx & 1023;
  const float* gr = g + (size_t)b * 4 * DEC_;
  float gi = gr[u]              + bih[u]              + bhh[u];
  float gf = gr[DEC_ + u]       + bih[DEC_ + u]       + bhh[DEC_ + u];
  float gg = gr[2 * DEC_ + u]   + bih[2 * DEC_ + u]   + bhh[2 * DEC_ + u];
  float go = gr[3 * DEC_ + u]   + bih[3 * DEC_ + u]   + bhh[3 * DEC_ + u];
  float cp = c[idx], hp = h[idx];
  float c2 = sigf(gf) * cp + sigf(gi) * tanhf(gg);
  float h2 = sigf(go) * tanhf(c2);
  float hn = ZONEOUT * hp + (1.0f - ZONEOUT) * h2;
  float cn = ZONEOUT * cp + (1.0f - ZONEOUT) * c2;
  h[idx] = hn; c[idx] = cn;
  dsta[(size_t)b * ldda + u] = (_Float16)hn;
  dstb[(size_t)b * lddb + u] = (_Float16)hn;
}

__global__ void k_writeout(const float* __restrict__ gout, const float* __restrict__ prob_b,
                           float* __restrict__ out_feat, float* __restrict__ out_logit, int s) {
  int i = blockIdx.x * blockDim.x + threadIdx.x;
  if (i >= B_ * 162) return;
  int b = i / 162, q = i % 162;
  if (q < 160) {
    int r = q / CCH, ch = q % CCH;
    out_feat[(size_t)b * CCH * 1000 + (size_t)ch * 1000 + s * 2 + r] = gout[(size_t)b * 192 + q];
  } else {
    int r = q - 160;
    out_logit[(size_t)b * 1000 + s * 2 + r] = gout[(size_t)b * 192 + 160 + r] + prob_b[r];
  }
}

// ---------------------------------------------------------------------------
static inline void launch_gemm(const _Float16* A, int lda, const _Float16* Bp,
                               float* C, int ldc, int M, int N, int K, hipStream_t st) {
  int waves = (M / 16) * (N / 64);
  int blocks = (waves * 32 + 255) / 256;
  gemm_wmma<<<blocks, 256, 0, st>>>(A, lda, Bp, C, ldc, M, N, K);
}

extern "C" void kernel_launch(void* const* d_in, const int* in_sizes, int n_in,
                              void* d_out, int out_size, void* d_ws, size_t ws_size,
                              hipStream_t stream) {
  const float* enc      = (const float*)d_in[0];
  const int*   text_len = (const int*)d_in[1];
  const float* ftarget  = (const float*)d_in[2];
  const float* w_enc    = (const float*)d_in[3];
  const float* b_enc    = (const float*)d_in[4];
  const float* w_dec    = (const float*)d_in[5];
  const float* w_att    = (const float*)d_in[6];
  const float* w_conv   = (const float*)d_in[7];
  const float* w_ww     = (const float*)d_in[8];
  const float* w_wb     = (const float*)d_in[9];
  const float* pw0      = (const float*)d_in[10];
  const float* pb0      = (const float*)d_in[11];
  const float* pw1      = (const float*)d_in[12];
  const float* pb1      = (const float*)d_in[13];
  const float* l0_wih   = (const float*)d_in[14];
  const float* l0_whh   = (const float*)d_in[15];
  const float* l0_bih   = (const float*)d_in[16];
  const float* l0_bhh   = (const float*)d_in[17];
  const float* l1_wih   = (const float*)d_in[18];
  const float* l1_whh   = (const float*)d_in[19];
  const float* l1_bih   = (const float*)d_in[20];
  const float* l1_bhh   = (const float*)d_in[21];
  const float* featw    = (const float*)d_in[22];
  const float* probw    = (const float*)d_in[23];
  const float* probb    = (const float*)d_in[24];

  float* dout = (float*)d_out;
  const size_t O1 = (size_t)B_ * CCH * 1000;       // logits
  const size_t O2 = O1 + (size_t)B_ * 1000;        // attention weights

  char* ws = (char*)d_ws;
  size_t off = 0;
  auto alloc = [&](size_t bytes) { size_t o = off; off = (off + bytes + 255) & ~(size_t)255; return o; };

  _Float16* Wcat0 = (_Float16*)(ws + alloc((size_t)4096 * 1536 * 2));
  _Float16* Wcat1 = (_Float16*)(ws + alloc((size_t)4096 * 2048 * 2));
  _Float16* Wdec  = (_Float16*)(ws + alloc((size_t)128 * 1024 * 2));
  _Float16* Wpre0 = (_Float16*)(ws + alloc((size_t)256 * 160 * 2));
  _Float16* Wpre1 = (_Float16*)(ws + alloc((size_t)256 * 256 * 2));
  _Float16* Wenc  = (_Float16*)(ws + alloc((size_t)128 * 256 * 2));
  _Float16* Wout  = (_Float16*)(ws + alloc((size_t)192 * 1280 * 2));
  _Float16* ench  = (_Float16*)(ws + alloc((size_t)B_ * TENC * ENC_ * 2));
  float* pmbuf    = (float*)(ws + alloc((size_t)B_ * TENC * HATT * 4));
  float* g0       = (float*)(ws + alloc((size_t)B_ * 4096 * 4));
  float* g1       = (float*)(ws + alloc((size_t)B_ * 4096 * 4));
  float* gdec     = (float*)(ws + alloc((size_t)B_ * 128 * 4));
  float* gpre0    = (float*)(ws + alloc((size_t)B_ * 256 * 4));
  float* gpre1    = (float*)(ws + alloc((size_t)B_ * 256 * 4));
  float* gout     = (float*)(ws + alloc((size_t)B_ * 192 * 4));
  float* h0s      = (float*)(ws + alloc((size_t)B_ * DEC_ * 4));
  float* c0s      = (float*)(ws + alloc((size_t)B_ * DEC_ * 4));
  float* h1s      = (float*)(ws + alloc((size_t)B_ * DEC_ * 4));
  float* c1s      = (float*)(ws + alloc((size_t)B_ * DEC_ * 4));
  float* attacc   = (float*)(ws + alloc((size_t)B_ * TENC * 4));
  _Float16* A0    = (_Float16*)(ws + alloc((size_t)B_ * 1536 * 2));  // [ac | p | h0]
  _Float16* A1    = (_Float16*)(ws + alloc((size_t)B_ * 2048 * 2));  // [h0 | h1]
  _Float16* Ahcs  = (_Float16*)(ws + alloc((size_t)B_ * 1280 * 2));  // [h1 | ac]
  _Float16* p0h   = (_Float16*)(ws + alloc((size_t)B_ * 256 * 2));
  _Float16* pin   = (_Float16*)(ws + alloc((size_t)B_ * 160 * 2));

  auto nb = [](size_t n) { return (unsigned)((n + 255) / 256); };

  // ---- one-time setup: pack weights (f16 fragment order), pm precompute ----
  k_pack_catK<<<nb((size_t)4096 * 1536), 256, 0, stream>>>(l0_wih, l0_whh, Wcat0, 4096, 512, 1536);
  k_pack_catK<<<nb((size_t)4096 * 2048), 256, 0, stream>>>(l1_wih, l1_whh, Wcat1, 4096, 1024, 2048);
  k_pack_catK<<<nb((size_t)128 * 1024), 256, 0, stream>>>(w_dec, w_dec, Wdec, 128, 1024, 1024);
  k_pack_catK<<<nb((size_t)256 * 160), 256, 0, stream>>>(pw0, pw0, Wpre0, 256, 160, 160);
  k_pack_catK<<<nb((size_t)256 * 256), 256, 0, stream>>>(pw1, pw1, Wpre1, 256, 256, 256);
  k_pack_catK<<<nb((size_t)128 * 256), 256, 0, stream>>>(w_enc, w_enc, Wenc, 128, 256, 256);
  k_pack_catN<<<nb((size_t)192 * 1280), 256, 0, stream>>>(featw, probw, Wout, 160, 2, 192, 1280);
  k_f32_to_f16<<<nb((size_t)B_ * TENC * ENC_), 256, 0, stream>>>(enc, ench, (size_t)B_ * TENC * ENC_);
  // pm = enc @ att_fc_enc_w.T  (bias added inside attention kernel)
  launch_gemm(ench, ENC_, Wenc, pmbuf, HATT, B_ * TENC, HATT, ENC_, stream);
  // zero recurrent state + activation buffers (re-done each call: deterministic)
  k_zero_f32<<<nb((size_t)B_ * DEC_), 256, 0, stream>>>(h0s, (size_t)B_ * DEC_);
  k_zero_f32<<<nb((size_t)B_ * DEC_), 256, 0, stream>>>(c0s, (size_t)B_ * DEC_);
  k_zero_f32<<<nb((size_t)B_ * DEC_), 256, 0, stream>>>(h1s, (size_t)B_ * DEC_);
  k_zero_f32<<<nb((size_t)B_ * DEC_), 256, 0, stream>>>(c1s, (size_t)B_ * DEC_);
  k_zero_f16<<<nb((size_t)B_ * 1536), 256, 0, stream>>>(A0, (size_t)B_ * 1536);
  k_zero_f16<<<nb((size_t)B_ * 2048), 256, 0, stream>>>(A1, (size_t)B_ * 2048);
  k_zero_f16<<<nb((size_t)B_ * 1280), 256, 0, stream>>>(Ahcs, (size_t)B_ * 1280);

  // ---- autoregressive loop: 500 teacher-forced steps ----
  for (int s = 0; s < SSTEPS; ++s) {
    // h0_prev @ att_fc_dec_w.T (h0 lives in A0 cols 512..1535)
    launch_gemm(A0 + 512, 1536, Wdec, gdec, 128, B_, 128, 1024, stream);
    k_attention<<<B_, 256, 0, stream>>>(enc, text_len, pmbuf, gdec, w_conv, w_att,
                                        w_ww, w_wb, b_enc, attacc, dout + O2, A0, Ahcs, s);
    k_prenet_in<<<nb((size_t)B_ * 160), 256, 0, stream>>>(ftarget, pin, s);
    launch_gemm(pin, 160, Wpre0, gpre0, 256, B_, 256, 160, stream);
    k_bias_relu_f16<<<nb((size_t)B_ * 256), 256, 0, stream>>>(gpre0, pb0, p0h, 256, 256);
    launch_gemm(p0h, 256, Wpre1, gpre1, 256, B_, 256, 256, stream);
    k_bias_relu_f16<<<nb((size_t)B_ * 256), 256, 0, stream>>>(gpre1, pb1, A0 + 256, 256, 1536);
    // LSTM0: [ac|p|h0] (64x1536) x [wih;whh]^T (1536x4096)
    launch_gemm(A0, 1536, Wcat0, g0, 4096, B_, 4096, 1536, stream);
    k_lstm_pw<<<nb((size_t)B_ * DEC_), 256, 0, stream>>>(g0, l0_bih, l0_bhh, h0s, c0s,
                                                         A0 + 512, 1536, A1, 2048);
    // LSTM1: [h0|h1] (64x2048) x [wih;whh]^T (2048x4096)
    launch_gemm(A1, 2048, Wcat1, g1, 4096, B_, 4096, 2048, stream);
    k_lstm_pw<<<nb((size_t)B_ * DEC_), 256, 0, stream>>>(g1, l1_bih, l1_bhh, h1s, c1s,
                                                         A1 + 1024, 2048, Ahcs, 1280);
    // Output projections: [h1|ac] (64x1280) x [feat;prob]^T (1280x192 padded)
    launch_gemm(Ahcs, 1280, Wout, gout, 192, B_, 192, 1280, stream);
    k_writeout<<<nb((size_t)B_ * 162), 256, 0, stream>>>(gout, probb, dout, dout + O1, s);
  }
}
